// NeuralJumpODE_56865366999503
// MI455X (gfx1250) — matrix operand order repaired and checked by reading, not compiled, not run on hardware
//
#include <hip/hip_runtime.h>
#include <math.h>

// ---------- problem constants ----------
#define BB  256
#define NN  1024
#define DXX 16
#define HH  64
#define DYY 8
#define MM  2
#define OUT1 (BB * NN * DYY * MM)   // offset of obs_pred_before in d_out

typedef __attribute__((ext_vector_type(16))) _Float16 v16h;
typedef __attribute__((ext_vector_type(8)))  float    v8f;

union V16HU { v16h v; uint4 q[2]; _Float16 h[16]; };

// LDS weight bank offsets (halves). All stored transposed: [n][k], K zero-padded.
#define OFF_JW1 0        // 64 x 32
#define OFF_JW2 2048     // 64 x 64
#define OFF_UW1 6144     // 64 x 64
#define OFF_UW2 10240    // 16 x 64  (rows >= 8 zero)
#define OFF_OW1 11264    // 64 x 96  (k >= 82 zero)
#define OFF_OW2 17408    // 64 x 64
#define SW_TOTAL 21504

// Bias bank offsets (floats)
#define BOFF_JB1 0
#define BOFF_JB2 64
#define BOFF_UB1 128
#define BOFF_OB1 192
#define BOFF_OB2 256
#define BOFF_UB2 320
#define SB_TOTAL 328

// Branchless tanh on the hardware transcendental pipe:
// tanh(x) = 1 - 2/(exp(2x)+1) = 1 - 2*rcp(exp2(2*log2e*x)+1).
// Saturates to +/-1 for large |x|; no EXEC divergence (keeps WMMA env clean).
static __device__ __forceinline__ float fast_tanh(float x) {
  float e = __builtin_amdgcn_exp2f(x * 2.885390081777927f);
  float r = __builtin_amdgcn_rcpf(e + 1.0f);
  return __builtin_fmaf(-2.0f, r, 1.0f);
}

static __device__ __forceinline__ v8f wmma_f16(v16h a, v16h b, v8f c) {
  // D = A(16x32 f16) * B(32x16 f16) + C(16x16 f32)
  return __builtin_amdgcn_wmma_f32_16x16x32_f16(false, a, false, b, (short)0, c,
                                                false, false);
}

// B fragment from transposed weight tile in LDS.
// Lane = output column (n), lanes 0-15 hold K 0-7 & K16-23 of the chunk, 16-31 hold K 8-15 & K24-31.
static __device__ __forceinline__ v16h load_bfrag(const _Float16* wt, int kpad,
                                                  int ctile, int kc, int lane) {
  int row  = ctile * 16 + (lane & 15);
  int half = lane >> 4;
  const _Float16* p = wt + row * kpad + kc * 32 + half * 8;
  V16HU f;
  f.q[0] = *(const uint4*)(p);
  f.q[1] = *(const uint4*)(p + 16);
  return f.v;
}

// A fragment from a 16x64 f16 activation tile in LDS (row-major, stride 64).
// Lanes 0-15: K0-7 & K16-23; lanes 16-31: K8-15 & K24-31 (per ISA A layout).
static __device__ __forceinline__ v16h load_afrag(const _Float16* tile, int kc,
                                                  int lane) {
  int row  = lane & 15;
  int half = lane >> 4;
  const _Float16* p = tile + row * 64 + kc * 32 + half * 8;
  V16HU f;
  f.q[0] = *(const uint4*)(p);
  f.q[1] = *(const uint4*)(p + 16);
  return f.v;
}

// Store a 16x16 f32 C fragment (as f16) into the activation tile at column tile c.
static __device__ __forceinline__ void store_ctile(_Float16* tile, int c, int lane,
                                                   v8f vals) {
  int col = c * 16 + (lane & 15);
  int rowbase = (lane >> 4) * 8;
#pragma unroll
  for (int r = 0; r < 8; ++r)
    tile[(rowbase + r) * 64 + col] = (_Float16)vals[r];
}

__global__ __launch_bounds__(256) void njode_kernel(
    const float* __restrict__ times, const float* __restrict__ values,
    const float* __restrict__ jW1, const float* __restrict__ jb1,
    const float* __restrict__ jW2, const float* __restrict__ jb2,
    const float* __restrict__ oW1, const float* __restrict__ ob1,
    const float* __restrict__ oW2, const float* __restrict__ ob2,
    const float* __restrict__ uW1, const float* __restrict__ ub1,
    const float* __restrict__ uW2, const float* __restrict__ ub2,
    float* __restrict__ out) {
  __shared__ __align__(16) _Float16 sW[SW_TOTAL];
  __shared__ __align__(16) _Float16 sTile[8][16 * 64];
  __shared__ float sDt[8][16];
  __shared__ float sB[SB_TOTAL];

  const int tid = threadIdx.x;
  const int m   = blockIdx.x & 1;

  // ---- cooperative weight staging: f32 global -> transposed f16 LDS ----
  for (int i = tid; i < 2048; i += 256) {           // jW1t 64x32
    int n = i >> 5, k = i & 31;
    sW[OFF_JW1 + i] = (k < DXX) ? (_Float16)jW1[(m * DXX + k) * HH + n]
                                : (_Float16)0.f;
  }
  for (int i = tid; i < 4096; i += 256) {           // jW2t 64x64
    int n = i >> 6, k = i & 63;
    sW[OFF_JW2 + i] = (_Float16)jW2[(m * HH + k) * HH + n];
  }
  for (int i = tid; i < 4096; i += 256) {           // uW1t 64x64
    int n = i >> 6, k = i & 63;
    sW[OFF_UW1 + i] = (_Float16)uW1[(m * HH + k) * HH + n];
  }
  for (int i = tid; i < 1024; i += 256) {           // uW2t 16x64 (pad rows)
    int n = i >> 6, k = i & 63;
    sW[OFF_UW2 + i] = (n < DYY) ? (_Float16)uW2[(m * HH + k) * DYY + n]
                                : (_Float16)0.f;
  }
  for (int i = tid; i < 6144; i += 256) {           // oW1t 64x96 (pad K)
    int n = i / 96, k = i % 96;
    sW[OFF_OW1 + i] = (k < HH + DXX + 2)
                          ? (_Float16)oW1[(m * (HH + DXX + 2) + k) * HH + n]
                          : (_Float16)0.f;
  }
  for (int i = tid; i < 4096; i += 256) {           // oW2t 64x64
    int n = i >> 6, k = i & 63;
    sW[OFF_OW2 + i] = (_Float16)oW2[(m * HH + k) * HH + n];
  }
  // ---- bias staging ----
  for (int i = tid; i < SB_TOTAL; i += 256) {
    float v;
    if      (i < 64)  v = jb1[m * HH + i];
    else if (i < 128) v = jb2[m * HH + (i - 64)];
    else if (i < 192) v = ub1[m * HH + (i - 128)];
    else if (i < 256) v = ob1[m * HH + (i - 192)];
    else if (i < 320) v = ob2[m * HH + (i - 256)];
    else              v = ub2[m * DYY + (i - 320)];
    sB[i] = v;
  }
  __syncthreads();

  const int lane = tid & 31;
  const int wave = tid >> 5;
  _Float16* myTile = sTile[wave];
  float*    myDt   = sDt[wave];

  const int row     = lane & 15;
  const int half    = lane >> 4;
  const int rowbase = half * 8;
  const int col     = lane & 15;

  // grid = 512 blocks (256 per m); 256 block-groups * 8 waves = 2048 waves per m;
  // 16384 tiles per m -> each wave loops over exactly 8 tiles. No divergence.
  const int gwave = (blockIdx.x >> 1) * 8 + wave;

  for (int tile = gwave; tile < (BB * NN / 16); tile += 2048) {
    const int b  = tile >> 6;                       // 64 tiles of 16 per batch row
    const int n0 = (tile & 63) << 4;
    const long tok0 = (long)b * NN + n0;

    // ---- A-frag of X: 16 tokens x DX(=16), K padded to 32 ----
    V16HU xa;
    {
      const float* xp = values + (tok0 + row) * DXX + half * 8;
#pragma unroll
      for (int i = 0; i < 8; ++i) xa.h[i] = (_Float16)xp[i];
#pragma unroll
      for (int i = 8; i < 16; ++i) xa.h[i] = (_Float16)0.f;
    }

    // ---- h1 = tanh(X @ jW1 + jb1) ----
#pragma unroll
    for (int c = 0; c < 4; ++c) {
      v8f acc = {};
      acc = wmma_f16(xa.v, load_bfrag(sW + OFF_JW1, 32, c, 0, lane), acc);
      float bias = sB[BOFF_JB1 + c * 16 + col];
      v8f t;
#pragma unroll
      for (int r = 0; r < 8; ++r) t[r] = fast_tanh(acc[r] + bias);
      store_ctile(myTile, c, lane, t);
    }
    v16h a0 = load_afrag(myTile, 0, lane);
    v16h a1 = load_afrag(myTile, 1, lane);

    // ---- h = tanh(h1 @ jW2 + jb2); keep f32 copy for Euler step ----
    v8f hC[4];
#pragma unroll
    for (int c = 0; c < 4; ++c) {
      v8f acc = {};
      acc = wmma_f16(a0, load_bfrag(sW + OFF_JW2, 64, c, 0, lane), acc);
      acc = wmma_f16(a1, load_bfrag(sW + OFF_JW2, 64, c, 1, lane), acc);
      float bias = sB[BOFF_JB2 + c * 16 + col];
#pragma unroll
      for (int r = 0; r < 8; ++r) hC[c][r] = fast_tanh(acc[r] + bias);
      store_ctile(myTile, c, lane, hC[c]);
    }
    v16h ha0 = load_afrag(myTile, 0, lane);
    v16h ha1 = load_afrag(myTile, 1, lane);

    // ---- obs_pred = out_net(h): u = tanh(h@uW1+ub1); y = u@uW2+ub2 ----
#pragma unroll
    for (int c = 0; c < 4; ++c) {
      v8f acc = {};
      acc = wmma_f16(ha0, load_bfrag(sW + OFF_UW1, 64, c, 0, lane), acc);
      acc = wmma_f16(ha1, load_bfrag(sW + OFF_UW1, 64, c, 1, lane), acc);
      float bias = sB[BOFF_UB1 + c * 16 + col];
      v8f t;
#pragma unroll
      for (int r = 0; r < 8; ++r) t[r] = fast_tanh(acc[r] + bias);
      store_ctile(myTile, c, lane, t);
    }
    {
      v16h ua0 = load_afrag(myTile, 0, lane);
      v16h ua1 = load_afrag(myTile, 1, lane);
      v8f acc = {};
      acc = wmma_f16(ua0, load_bfrag(sW + OFF_UW2, 64, 0, 0, lane), acc);
      acc = wmma_f16(ua1, load_bfrag(sW + OFF_UW2, 64, 0, 1, lane), acc);
      if (col < DYY) {
        float bias = sB[BOFF_UB2 + col];
#pragma unroll
        for (int r = 0; r < 8; ++r) {
          long n = tok0 + rowbase + r;
          out[(n * DYY + col) * MM + m] = acc[r] + bias;
        }
      }
    }

    // ---- before branch: inp = [h, x, t_rel, t_el] (82 -> K pad 96) ----
    V16HU ca;
    {
      long tIdx = tok0 + row;
      const float* xp = values + tIdx * DXX + half * 8;
#pragma unroll
      for (int i = 0; i < 8; ++i) ca.h[i] = (_Float16)xp[i];
#pragma unroll
      for (int i = 8; i < 16; ++i) ca.h[i] = (_Float16)0.f;
      if (half == 0) {         // lanes 0-15 carry K 80..87 of inp: t_rel, t_el
        float tr = times[tIdx];
        long nx = tIdx + 1;
        if (nx > (long)BB * NN - 1) nx = (long)BB * NN - 1;
        float te = times[nx] - tr;
        ca.h[8] = (_Float16)tr;
        ca.h[9] = (_Float16)te;
        myDt[row] = te;        // dt for the Euler step, indexed by token row
      }
    }

    // d1 = tanh(inp @ oW1 + ob1)
#pragma unroll
    for (int c = 0; c < 4; ++c) {
      v8f acc = {};
      acc = wmma_f16(ha0, load_bfrag(sW + OFF_OW1, 96, c, 0, lane), acc);
      acc = wmma_f16(ha1, load_bfrag(sW + OFF_OW1, 96, c, 1, lane), acc);
      acc = wmma_f16(ca.v, load_bfrag(sW + OFF_OW1, 96, c, 2, lane), acc);
      float bias = sB[BOFF_OB1 + c * 16 + col];
      v8f t;
#pragma unroll
      for (int r = 0; r < 8; ++r) t[r] = fast_tanh(acc[r] + bias);
      store_ctile(myTile, c, lane, t);
    }
    v16h da0 = load_afrag(myTile, 0, lane);
    v16h da1 = load_afrag(myTile, 1, lane);

    float dtv[8];
#pragma unroll
    for (int r = 0; r < 8; ++r) dtv[r] = myDt[rowbase + r];

    // dh = d1 @ oW2 + ob2 ; h_minus = h + dt*dh
#pragma unroll
    for (int c = 0; c < 4; ++c) {
      v8f acc = {};
      acc = wmma_f16(da0, load_bfrag(sW + OFF_OW2, 64, c, 0, lane), acc);
      acc = wmma_f16(da1, load_bfrag(sW + OFF_OW2, 64, c, 1, lane), acc);
      float bias = sB[BOFF_OB2 + c * 16 + col];
      v8f hm;
#pragma unroll
      for (int r = 0; r < 8; ++r) hm[r] = hC[c][r] + dtv[r] * (acc[r] + bias);
      store_ctile(myTile, c, lane, hm);
    }
    v16h hma0 = load_afrag(myTile, 0, lane);
    v16h hma1 = load_afrag(myTile, 1, lane);

    // yb = out_net(h_minus), written to rows n0+1..n0+16 of obs_pred_before
#pragma unroll
    for (int c = 0; c < 4; ++c) {
      v8f acc = {};
      acc = wmma_f16(hma0, load_bfrag(sW + OFF_UW1, 64, c, 0, lane), acc);
      acc = wmma_f16(hma1, load_bfrag(sW + OFF_UW1, 64, c, 1, lane), acc);
      float bias = sB[BOFF_UB1 + c * 16 + col];
      v8f t;
#pragma unroll
      for (int r = 0; r < 8; ++r) t[r] = fast_tanh(acc[r] + bias);
      store_ctile(myTile, c, lane, t);
    }
    {
      v16h ua0 = load_afrag(myTile, 0, lane);
      v16h ua1 = load_afrag(myTile, 1, lane);
      v8f acc = {};
      acc = wmma_f16(ua0, load_bfrag(sW + OFF_UW2, 64, 0, 0, lane), acc);
      acc = wmma_f16(ua1, load_bfrag(sW + OFF_UW2, 64, 0, 1, lane), acc);
      if (col < DYY) {
        float bias = sB[BOFF_UB2 + col];
#pragma unroll
        for (int r = 0; r < 8; ++r) {
          int nl = n0 + rowbase + r + 1;
          if (nl < NN)
            out[OUT1 + (((long)b * NN + nl) * DYY + col) * MM + m] =
                acc[r] + bias;
        }
      }
    }

    // obs_pred_before row 0 is zero
    if (n0 == 0 && lane < DYY)
      out[OUT1 + (((long)b * NN) * DYY + lane) * MM + m] = 0.f;
  }
}

extern "C" void kernel_launch(void* const* d_in, const int* in_sizes, int n_in,
                              void* d_out, int out_size, void* d_ws, size_t ws_size,
                              hipStream_t stream) {
  (void)in_sizes; (void)n_in; (void)out_size; (void)d_ws; (void)ws_size;
  const float* times = (const float*)d_in[0];
  const float* values = (const float*)d_in[1];
  const float* jW1 = (const float*)d_in[2];
  const float* jb1 = (const float*)d_in[3];
  const float* jW2 = (const float*)d_in[4];
  const float* jb2 = (const float*)d_in[5];
  const float* oW1 = (const float*)d_in[6];
  const float* ob1 = (const float*)d_in[7];
  const float* oW2 = (const float*)d_in[8];
  const float* ob2 = (const float*)d_in[9];
  const float* uW1 = (const float*)d_in[10];
  const float* ub1 = (const float*)d_in[11];
  const float* uW2 = (const float*)d_in[12];
  const float* ub2 = (const float*)d_in[13];
  float* out = (float*)d_out;

  // 512 blocks x 256 threads: 2048 waves per m, 8 tile-iterations per wave.
  dim3 grid(512), block(256);
  njode_kernel<<<grid, block, 0, stream>>>(times, values, jW1, jb1, jW2, jb2,
                                           oW1, ob1, oW2, ob2, uW1, ub1, uW2,
                                           ub2, out);
}
